// MixHopNet_61942018342913
// MI455X (gfx1250) — compile-verified, matrix operand order checked by reading, not run on hardware
//
#include <hip/hip_runtime.h>
#include <hip/hip_bf16.h>

#define N_NODES 100000
#define CHAN    64
#define N_OUT   40
#define OUT_PAD 48
#define CONCAT  192

typedef __attribute__((ext_vector_type(16))) __bf16          v16bf;
typedef __attribute__((ext_vector_type(8)))  __bf16          v8bf;
typedef __attribute__((ext_vector_type(8)))  float           v8f;
typedef __attribute__((ext_vector_type(4)))  float           f32x4;

// ---------------- workspace init: deg=1 (self loop), h1=h2=0 ----------------
__global__ void init_ws(float* __restrict__ deg, float* __restrict__ h1,
                        float* __restrict__ h2) {
    long long i = (long long)blockIdx.x * blockDim.x + threadIdx.x;
    if (i < (long long)N_NODES * CHAN) { h1[i] = 0.0f; h2[i] = 0.0f; }
    if (i < N_NODES) deg[i] = 1.0f;
}

// ---------------- degree accumulation over edge targets ---------------------
__global__ void deg_kernel(const int* __restrict__ col, float* __restrict__ deg, int E) {
    int e = blockIdx.x * blockDim.x + threadIdx.x;
    if (e < E) unsafeAtomicAdd(&deg[col[e]], 1.0f);
}

__global__ void dinv_kernel(const float* __restrict__ deg, float* __restrict__ dinv) {
    int i = blockIdx.x * blockDim.x + threadIdx.x;
    if (i < N_NODES) dinv[i] = rsqrtf(deg[i]);   // deg >= 1 always
}

// ---------------- SpMM: dst[t] += dinv[s]*dinv[t]*src[s] (edges + self) -----
__global__ void prop_kernel(const float* __restrict__ src, float* __restrict__ dst,
                            const int* __restrict__ row, const int* __restrict__ col,
                            const float* __restrict__ dinv, int E) {
    long long gid  = (long long)blockIdx.x * blockDim.x + threadIdx.x;
    long long item = gid >> 6;
    int       j    = (int)(gid & 63);
    if (item >= (long long)E + N_NODES) return;
    int s, t;
    if (item < E) { s = row[item]; t = col[item]; }
    else          { s = t = (int)(item - E); }
    float w = dinv[s] * dinv[t];
    unsafeAtomicAdd(&dst[(long long)t * CHAN + j], w * src[(long long)s * CHAN + j]);
}

// ---------------- fragment helpers (CDNA5 wave32 WMMA layouts) --------------
// A (16x32 bf16): lane l: M = l&15, half = l>>4; elems 0..7 -> K = kk*32+half*8+0..7,
//                 elems 8..15 -> K = kk*32+16+half*8+0..7. Same shape for B from a
//                 column-major (N-major) store: contiguous K chunks of 8.
__device__ __forceinline__ v16bf ld_bf16_frag_lds(const __bf16* q) {
    v8bf c0 = *(const v8bf*)q;
    v8bf c1 = *(const v8bf*)(q + 16);
    return __builtin_shufflevector(c0, c1, 0, 1, 2, 3, 4, 5, 6, 7,
                                   8, 9, 10, 11, 12, 13, 14, 15);
}

__device__ __forceinline__ v16bf ld_a_frag_f32(const float* pr) {
    f32x4 f0 = *(const f32x4*)pr;
    f32x4 f1 = *(const f32x4*)(pr + 4);
    f32x4 f2 = *(const f32x4*)(pr + 16);
    f32x4 f3 = *(const f32x4*)(pr + 20);
    v16bf f;
#pragma unroll
    for (int i = 0; i < 4; ++i) {
        f[i]      = (__bf16)f0[i];
        f[4 + i]  = (__bf16)f1[i];
        f[8 + i]  = (__bf16)f2[i];
        f[12 + i] = (__bf16)f3[i];
    }
    return f;
}

// One hop GEMM: 16x64 tile = A(16x64) x W(64x64), bias+ReLU, stage bf16 into Hw.
__device__ __forceinline__ void hop_gemm(const float* __restrict__ src,
                                         const __bf16* __restrict__ WtSp,
                                         const float* __restrict__ biasSp,
                                         __bf16* __restrict__ Hw,
                                         int base, int p, int hs, int lm) {
    v8f acc[4];
#pragma unroll
    for (int n = 0; n < 4; ++n) acc[n] = (v8f)0.0f;

#pragma unroll
    for (int kk = 0; kk < 2; ++kk) {
        v16bf a = ld_a_frag_f32(src + (long long)(base + lm) * CHAN + kk * 32 + hs * 8);
#pragma unroll
        for (int n = 0; n < 4; ++n) {
            v16bf bfr = ld_bf16_frag_lds(WtSp + (n * 16 + lm) * CHAN + kk * 32 + hs * 8);
            acc[n] = __builtin_amdgcn_wmma_f32_16x16x32_bf16(
                false, a, false, bfr, (short)0, acc[n], false, false);
        }
    }
    // C layout: vgpr r -> M = r + hs*8, N = lm. Fuse bias + ReLU, stage to LDS.
#pragma unroll
    for (int n = 0; n < 4; ++n) {
        float bias = biasSp[n * 16 + lm];
#pragma unroll
        for (int r = 0; r < 8; ++r) {
            float v = acc[n][r] + bias;
            v = fmaxf(v, 0.0f);
            Hw[(r + hs * 8) * CONCAT + p * CHAN + n * 16 + lm] = (__bf16)v;
        }
    }
}

// ---------------- fused MixHop GEMM + head -----------------------------------
__global__ __launch_bounds__(64) void mixhop_gemm(
    const float* __restrict__ x,  const float* __restrict__ h1, const float* __restrict__ h2,
    const float* __restrict__ W0, const float* __restrict__ b0,
    const float* __restrict__ W1, const float* __restrict__ b1,
    const float* __restrict__ W2, const float* __restrict__ b2,
    const float* __restrict__ Wl, const float* __restrict__ bl,
    float* __restrict__ out) {
    __shared__ __align__(16) __bf16 WtS[3 * CHAN * CHAN];   // [p][n][k] transposed
    __shared__ __align__(16) __bf16 WlTS[OUT_PAD * CONCAT]; // [n][k] transposed
    __shared__ __align__(16) __bf16 Hst[2 * 16 * CONCAT];   // per-wave concat staging
    __shared__ float biasS[3 * CHAN];
    __shared__ float blS[OUT_PAD];

    int tid = threadIdx.x;
    // --- weights -> LDS, transposed (N-major) + bf16 ---
    for (int idx = tid; idx < CHAN * CHAN; idx += 64) {
        int n = idx >> 6, k = idx & 63;
        WtS[0 * 4096 + idx] = (__bf16)W0[k * CHAN + n];
        WtS[1 * 4096 + idx] = (__bf16)W1[k * CHAN + n];
        WtS[2 * 4096 + idx] = (__bf16)W2[k * CHAN + n];
    }
    for (int idx = tid; idx < OUT_PAD * CONCAT; idx += 64) {
        int n = idx / CONCAT, k = idx - n * CONCAT;
        WlTS[idx] = (n < N_OUT) ? (__bf16)Wl[k * N_OUT + n] : (__bf16)0.0f;
    }
    if (tid < CHAN) {
        biasS[tid]            = b0[tid];
        biasS[CHAN + tid]     = b1[tid];
        biasS[2 * CHAN + tid] = b2[tid];
    }
    if (tid < OUT_PAD) blS[tid] = (tid < N_OUT) ? bl[tid] : 0.0f;
    __syncthreads();

    const int wave = tid >> 5;
    const int lane = tid & 31;
    const int hs   = lane >> 4;
    const int lm   = lane & 15;
    __bf16* Hw = Hst + wave * 16 * CONCAT;
    const int nTiles = N_NODES / 16;   // 6250, exact

    for (int tile = blockIdx.x * 2 + wave; tile < nTiles; tile += (int)gridDim.x * 2) {
        int base = tile * 16;

        hop_gemm(x,  WtS,            biasS,            Hw, base, 0, hs, lm);
        hop_gemm(h1, WtS + 4096,     biasS + CHAN,     Hw, base, 1, hs, lm);
        hop_gemm(h2, WtS + 2 * 4096, biasS + 2 * CHAN, Hw, base, 2, hs, lm);

        // wave-internal LDS ordering: H writes visible before cross-lane reads
        __builtin_amdgcn_wave_barrier();
        asm volatile("s_wait_dscnt 0" ::: "memory");
        __builtin_amdgcn_wave_barrier();

        // head: (16x192) @ (192x48), bias, store 40 valid cols
        v8f acc2[3];
#pragma unroll
        for (int nt = 0; nt < 3; ++nt) acc2[nt] = (v8f)0.0f;
#pragma unroll
        for (int kk = 0; kk < 6; ++kk) {
            v16bf a = ld_bf16_frag_lds(Hw + lm * CONCAT + kk * 32 + hs * 8);
#pragma unroll
            for (int nt = 0; nt < 3; ++nt) {
                v16bf bfr = ld_bf16_frag_lds(WlTS + (nt * 16 + lm) * CONCAT + kk * 32 + hs * 8);
                acc2[nt] = __builtin_amdgcn_wmma_f32_16x16x32_bf16(
                    false, a, false, bfr, (short)0, acc2[nt], false, false);
            }
        }
#pragma unroll
        for (int nt = 0; nt < 3; ++nt) {
            int c = nt * 16 + lm;
            if (c < N_OUT) {
                float bias = blS[c];
#pragma unroll
                for (int r = 0; r < 8; ++r) {
                    out[(long long)(base + r + hs * 8) * N_OUT + c] = acc2[nt][r] + bias;
                }
            }
        }
        __builtin_amdgcn_wave_barrier();
        asm volatile("s_wait_dscnt 0" ::: "memory");  // WAR safety on Hw reuse
    }
}

// ---------------------------------------------------------------------------
extern "C" void kernel_launch(void* const* d_in, const int* in_sizes, int n_in,
                              void* d_out, int out_size, void* d_ws, size_t ws_size,
                              hipStream_t stream) {
    const float* x  = (const float*)d_in[0];
    const int*   ei = (const int*)d_in[1];
    const float* W0 = (const float*)d_in[2];
    const float* b0 = (const float*)d_in[3];
    const float* W1 = (const float*)d_in[4];
    const float* b1 = (const float*)d_in[5];
    const float* W2 = (const float*)d_in[6];
    const float* b2 = (const float*)d_in[7];
    const float* Wl = (const float*)d_in[8];
    const float* bl = (const float*)d_in[9];
    float* out = (float*)d_out;

    const int E   = in_sizes[1] / 2;
    const int* row = ei;
    const int* col = ei + E;

    float* ws   = (float*)d_ws;
    float* deg  = ws;
    float* dinv = ws + N_NODES;
    float* h1   = ws + 2 * N_NODES;
    float* h2   = ws + 2 * N_NODES + (long long)N_NODES * CHAN;

    {   // init
        long long tot = (long long)N_NODES * CHAN;
        int blocks = (int)((tot + 255) / 256);
        init_ws<<<blocks, 256, 0, stream>>>(deg, h1, h2);
    }
    deg_kernel<<<(E + 255) / 256, 256, 0, stream>>>(col, deg, E);
    dinv_kernel<<<(N_NODES + 255) / 256, 256, 0, stream>>>(deg, dinv);
    {
        long long tot = ((long long)E + N_NODES) * 64;
        int blocks = (int)((tot + 255) / 256);
        prop_kernel<<<blocks, 256, 0, stream>>>(x,  h1, row, col, dinv, E);  // h1 = A_hat x
        prop_kernel<<<blocks, 256, 0, stream>>>(h1, h2, row, col, dinv, E);  // h2 = A_hat h1
    }
    mixhop_gemm<<<512, 64, 0, stream>>>(x, h1, h2, W0, b0, W1, b1, W2, b2, Wl, bl, out);
}
